// ResRGCNLitmusEmbedder_75076028334433
// MI455X (gfx1250) — matrix operand order repaired and checked by reading, not compiled, not run on hardware
//
#include <hip/hip_runtime.h>

// ---------------- problem constants ----------------
#define N_NODES 100000
#define N_EDGES 1250000
#define F_IN    32
#define H_DIM   64
#define R_REL   8
#define G_GR    256
#define EMB_DIM 16
#define COLS    576          // 9 * 64 : [root | W_0 .. W_7] outputs
#define EPS     1e-5f

typedef __attribute__((ext_vector_type(16))) _Float16 v16h;
typedef __attribute__((ext_vector_type(8)))  float    v8f;

// ---------------- tiny utility kernels ----------------
__global__ void k_zero_f(float* p, int n) {
    int i = blockIdx.x * blockDim.x + threadIdx.x;
    if (i < n) p[i] = 0.f;
}
__global__ void k_zero_i(int* p, int n) {
    int i = blockIdx.x * blockDim.x + threadIdx.x;
    if (i < n) p[i] = 0;
}
__global__ void k_f32_to_f16(const float* __restrict__ a, _Float16* __restrict__ b, int n) {
    int i = blockIdx.x * blockDim.x + threadIdx.x;
    if (i < n) b[i] = (_Float16)a[i];
}

// ---------------- edge preprocessing: cnt[dst][rel], per-edge weight ----------------
__global__ void k_count(const int* __restrict__ dst, const int* __restrict__ et,
                        int* __restrict__ cnt) {
    int e = blockIdx.x * blockDim.x + threadIdx.x;
    if (e < N_EDGES) atomicAdd(&cnt[dst[e] * R_REL + et[e]], 1);
}
__global__ void k_edge_w(const int* __restrict__ dst, const int* __restrict__ et,
                         const int* __restrict__ cnt, float* __restrict__ ew) {
    int e = blockIdx.x * blockDim.x + threadIdx.x;
    if (e < N_EDGES) {
        int c = cnt[dst[e] * R_REL + et[e]];
        ew[e] = 1.0f / (float)(c > 1 ? c : 1);
    }
}

// ---------------- build concatenated weight, transposed, f16: wT[576][K] ----------------
// col n: n<64 -> root[k][n]; else r=(n-64)/64, c=(n-64)%64 -> W[r][k][c]
__global__ void k_build_wT(const float* __restrict__ root, const float* __restrict__ W,
                           _Float16* __restrict__ wT, int K) {
    int i = blockIdx.x * blockDim.x + threadIdx.x;
    if (i >= COLS * K) return;
    int n = i / K, k = i % K;
    float v;
    if (n < H_DIM) v = root[k * H_DIM + n];
    else {
        int r = (n - H_DIM) >> 6, c = (n - H_DIM) & 63;
        v = W[(size_t)r * K * H_DIM + k * H_DIM + c];
    }
    wT[(size_t)n * K + k] = (_Float16)v;
}

// ---------------- WMMA GEMM: h[N,576] = x[N,K] @ Wcat[K,576], f16 in / f32 acc ----------
// grid = (N/16, 36/4), block = 128 (4 waves); wave w -> col tile blockIdx.y*4+w
__global__ void __launch_bounds__(128)
k_gemm_wmma(const _Float16* __restrict__ xb, const _Float16* __restrict__ wT,
            float* __restrict__ h, int K) {
    int wave = threadIdx.x >> 5;
    int lane = threadIdx.x & 31;
    int mt = blockIdx.x;
    int ct = blockIdx.y * 4 + wave;
    int l15 = lane & 15;
    int hi  = lane >> 4;

    v8f c = {};
    // A 16x32 f16 layout: lane(l15) = row M, hi selects K groups {0..7,16..23} vs {8..15,24..31}
    const _Float16* arow = xb + (size_t)(mt * 16 + l15) * K + hi * 8;
    // B 32x16 layout: lane(l15) = col N, hi selects K groups {0..7,16..23} vs {8..15,24..31}; wT stores K contiguously
    const _Float16* brow = wT + (size_t)(ct * 16 + l15) * K + hi * 8;

    for (int kk = 0; kk < K; kk += 32) {
        v16h a, b;
#pragma unroll
        for (int j = 0; j < 8; ++j) {
            a[j]     = arow[kk + j];        // K = kk + hi*8 + j
            a[8 + j] = arow[kk + 16 + j];   // K = kk + hi*8 + 16 + j
        }
#pragma unroll
        for (int j = 0; j < 8; ++j) { b[j] = brow[kk + j]; b[8 + j] = brow[kk + 16 + j]; }   // K = kk + hi*8 + j, kk + hi*8 + 16 + j
        c = __builtin_amdgcn_wmma_f32_16x16x32_f16(false, a, false, b,
                                                   (short)0, c, false, false);
    }
    // C/D layout: element (M = v + 8*hi, N = l15) in vgpr v
    float* out = h + (size_t)(mt * 16 + hi * 8) * COLS + ct * 16 + l15;
#pragma unroll
    for (int v = 0; v < 8; ++v) out[(size_t)v * COLS] = c[v];
}

// ---------------- acc[n][j] = root-part + bias ----------------
__global__ void k_acc_init(const float* __restrict__ h, const float* __restrict__ bias,
                           float* __restrict__ acc) {
    int i = blockIdx.x * blockDim.x + threadIdx.x;
    if (i >= N_NODES * H_DIM) return;
    int n = i >> 6, j = i & 63;
    acc[i] = h[(size_t)n * COLS + j] + bias[j];
}

// ---------------- edge scatter: acc[dst] += ew[e] * h[src, rel-block] -----------------
// one wave per edge, 2 columns per lane
__global__ void k_scatter(const float* __restrict__ h, const int* __restrict__ src,
                          const int* __restrict__ dst, const int* __restrict__ et,
                          const float* __restrict__ ew, float* __restrict__ acc) {
    long long t = (long long)blockIdx.x * blockDim.x + threadIdx.x;
    int e = (int)(t >> 5);
    if (e >= N_EDGES) return;
    int lane = (int)(t & 31);
    int s = src[e], d = dst[e], r = et[e];
    float w = ew[e];
    const float* hp = h + (size_t)s * COLS + H_DIM + r * H_DIM + lane * 2;
    float* ap = acc + (size_t)d * H_DIM + lane * 2;
    atomicAdd(ap,     hp[0] * w);
    atomicAdd(ap + 1, hp[1] * w);
}

// ---------------- BN column stats (two-pass) ----------------
__global__ void k_colstats(const float* __restrict__ acc, float* __restrict__ sum,
                           float* __restrict__ sq) {
    int tid = blockIdx.x * blockDim.x + threadIdx.x;
    int col = tid & 63;
    int rowStart  = tid >> 6;
    int rowStride = (gridDim.x * blockDim.x) >> 6;
    float s = 0.f, q = 0.f;
    for (int n = rowStart; n < N_NODES; n += rowStride) {
        float v = acc[(size_t)n * H_DIM + col];
        s += v; q += v * v;
    }
    atomicAdd(&sum[col], s);
    atomicAdd(&sq[col], q);
}
__global__ void k_bn_coef(const float* __restrict__ sum, const float* __restrict__ sq,
                          const float* __restrict__ g, const float* __restrict__ b,
                          float* __restrict__ scale, float* __restrict__ shift) {
    int j = threadIdx.x;
    if (j >= H_DIM) return;
    float mu  = sum[j] / (float)N_NODES;
    float var = sq[j] / (float)N_NODES - mu * mu;
    float sc  = g[j] * rsqrtf(var + EPS);
    scale[j] = sc;
    shift[j] = b[j] - mu * sc;
}
__global__ void k_bn_relu(const float* __restrict__ acc, const float* __restrict__ scale,
                          const float* __restrict__ shift, float* __restrict__ xo,
                          _Float16* __restrict__ xb) {
    int i = blockIdx.x * blockDim.x + threadIdx.x;
    if (i >= N_NODES * H_DIM) return;
    int j = i & 63;
    float v = acc[i] * scale[j] + shift[j];
    v = v > 0.f ? v : 0.f;
    xo[i] = v;
    xb[i] = (_Float16)v;    // next layer's f16 GEMM input
}

// ---------------- graph pooling (sum + max + count) ----------------
__global__ void k_pool(const float* __restrict__ x1, const float* __restrict__ x2,
                       const float* __restrict__ x3, const int* __restrict__ batch,
                       float* __restrict__ psum, unsigned int* __restrict__ pmax,
                       float* __restrict__ pcnt) {
    long long t = (long long)blockIdx.x * blockDim.x + threadIdx.x;
    if (t >= (long long)N_NODES * 192) return;
    int n = (int)(t / 192), c = (int)(t % 192);
    const float* srcp = (c < 64) ? x1 : ((c < 128) ? x2 : x3);
    float v = srcp[(size_t)n * 64 + (c & 63)];
    int g = batch[n];
    atomicAdd(&psum[g * 192 + c], v);
    atomicMax(&pmax[g * 192 + c], __float_as_uint(v));   // v >= 0 post-ReLU: monotone
    if (c == 0) atomicAdd(&pcnt[g], 1.f);
}
__global__ void k_xgraph(const float* __restrict__ psum, const unsigned int* __restrict__ pmax,
                         const float* __restrict__ pcnt, float* __restrict__ xg) {
    int i = blockIdx.x * blockDim.x + threadIdx.x;
    if (i >= G_GR * 384) return;
    int g = i / 384, c = i % 384;
    float v;
    if (c < 192) v = psum[g * 192 + c] / fmaxf(pcnt[g], 1.f);
    else         v = __uint_as_float(pmax[g * 192 + (c - 192)]);
    xg[i] = v;
}

// ---------------- small MLP head ----------------
__global__ void k_fc1(const float* __restrict__ xg, const float* __restrict__ w,
                      const float* __restrict__ b, float* __restrict__ out) {
    int i = blockIdx.x * blockDim.x + threadIdx.x;
    if (i >= G_GR * 128) return;
    int g = i >> 7, o = i & 127;
    float s = b[o];
    for (int k = 0; k < 384; ++k) s += xg[g * 384 + k] * w[k * 128 + o];
    out[i] = s;
}
// BN over G rows + leaky; one block (256 threads == G) per column
__global__ void k_fcbn_leaky(const float* __restrict__ in, const float* __restrict__ g_,
                             const float* __restrict__ b_, float* __restrict__ out) {
    __shared__ float s1[256], s2[256];
    int c = blockIdx.x, t = threadIdx.x;
    float v = in[t * 128 + c];
    s1[t] = v; s2[t] = v * v;
    __syncthreads();
    for (int s = 128; s > 0; s >>= 1) {
        if (t < s) { s1[t] += s1[t + s]; s2[t] += s2[t + s]; }
        __syncthreads();
    }
    float mu  = s1[0] / (float)G_GR;
    float var = s2[0] / (float)G_GR - mu * mu;
    float z = g_[c] * (v - mu) * rsqrtf(var + EPS) + b_[c];
    out[t * 128 + c] = z > 0.f ? z : 0.2f * z;
}
__global__ void k_fc2(const float* __restrict__ h, const float* __restrict__ w,
                      const float* __restrict__ b, float* __restrict__ emb) {
    int i = blockIdx.x * blockDim.x + threadIdx.x;
    if (i >= G_GR * EMB_DIM) return;
    int g = i >> 4, o = i & 15;
    float s = b[o];
    for (int k = 0; k < 128; ++k) s += h[g * 128 + k] * w[k * EMB_DIM + o];
    emb[i] = s;
}
__global__ void k_dec(const float* __restrict__ emb, const float* __restrict__ w1,
                      const float* __restrict__ b1, const float* __restrict__ w2,
                      const float* __restrict__ b2, float* __restrict__ pred) {
    int g = blockIdx.x * blockDim.x + threadIdx.x;
    if (g >= G_GR) return;
    float e[EMB_DIM];
    for (int i = 0; i < EMB_DIM; ++i) e[i] = emb[g * EMB_DIM + i];
    float acc = b2[0];
    for (int j = 0; j < 64; ++j) {
        float t = b1[j];
        for (int i = 0; i < EMB_DIM; ++i) t += e[i] * w1[i * 64 + j];
        t = t > 0.f ? t : 0.2f * t;
        acc += t * w2[j];
    }
    pred[g] = acc;
}

// ---------------- host orchestration ----------------
extern "C" void kernel_launch(void* const* d_in, const int* in_sizes, int n_in,
                              void* d_out, int out_size, void* d_ws, size_t ws_size,
                              hipStream_t stream) {
    const float* x      = (const float*)d_in[0];
    const int*   ei     = (const int*)d_in[1];
    const int*   src    = ei;
    const int*   dst    = ei + N_EDGES;
    const int*   et     = (const int*)d_in[2];
    const int*   batch  = (const int*)d_in[3];
    const float* conv_w[3]    = {(const float*)d_in[4],  (const float*)d_in[9],  (const float*)d_in[14]};
    const float* conv_root[3] = {(const float*)d_in[5],  (const float*)d_in[10], (const float*)d_in[15]};
    const float* conv_b[3]    = {(const float*)d_in[6],  (const float*)d_in[11], (const float*)d_in[16]};
    const float* bn_g[3]      = {(const float*)d_in[7],  (const float*)d_in[12], (const float*)d_in[17]};
    const float* bn_b[3]      = {(const float*)d_in[8],  (const float*)d_in[13], (const float*)d_in[18]};
    const float* fc1_w  = (const float*)d_in[19];
    const float* fc1_b  = (const float*)d_in[20];
    const float* fcbn_g = (const float*)d_in[21];
    const float* fcbn_b = (const float*)d_in[22];
    const float* fc2_w  = (const float*)d_in[23];
    const float* fc2_b  = (const float*)d_in[24];
    const float* dec1_w = (const float*)d_in[25];
    const float* dec1_b = (const float*)d_in[26];
    const float* dec2_w = (const float*)d_in[27];
    const float* dec2_b = (const float*)d_in[28];

    float* pred_out = (float*)d_out;            // [G,1]
    float* emb_out  = (float*)d_out + G_GR;     // [G,16]

    // workspace carve-up
    char* ws = (char*)d_ws;
    size_t off = 0;
    auto alloc = [&](size_t bytes) -> void* {
        void* p = ws + off;
        off = (off + bytes + 255) & ~(size_t)255;
        return p;
    };
    int*       cnt_nr = (int*)alloc((size_t)N_NODES * R_REL * 4);
    float*     ew     = (float*)alloc((size_t)N_EDGES * 4);
    _Float16*  xb     = (_Float16*)alloc((size_t)N_NODES * H_DIM * 2);
    _Float16*  wT     = (_Float16*)alloc((size_t)COLS * H_DIM * 2);
    float*     h      = (float*)alloc((size_t)N_NODES * COLS * 4);     // 230.4 MB
    float*     acc    = (float*)alloc((size_t)N_NODES * H_DIM * 4);
    float*     xl[3];
    xl[0] = (float*)alloc((size_t)N_NODES * H_DIM * 4);
    xl[1] = (float*)alloc((size_t)N_NODES * H_DIM * 4);
    xl[2] = (float*)alloc((size_t)N_NODES * H_DIM * 4);
    float*     stats  = (float*)alloc(4 * H_DIM * 4);   // [sum|sq|scale|shift]
    float*     poolb  = (float*)alloc((size_t)(G_GR * 192 * 2 + G_GR) * 4);
    float*        psum = poolb;
    unsigned int* pmax = (unsigned int*)(poolb + G_GR * 192);
    float*        pcnt = poolb + 2 * G_GR * 192;
    float*     xg     = (float*)alloc((size_t)G_GR * 384 * 4);
    float*     fcpre  = (float*)alloc((size_t)G_GR * 128 * 4);
    float*     fcb    = (float*)alloc((size_t)G_GR * 128 * 4);

    auto nb = [](long long n, int b) { return (unsigned)((n + b - 1) / b); };

    // --- edge structure preprocessing (shared across layers) ---
    k_zero_i<<<nb((long long)N_NODES * R_REL, 256), 256, 0, stream>>>(cnt_nr, N_NODES * R_REL);
    k_count<<<nb(N_EDGES, 256), 256, 0, stream>>>(dst, et, cnt_nr);
    k_edge_w<<<nb(N_EDGES, 256), 256, 0, stream>>>(dst, et, cnt_nr, ew);

    // --- layer-1 f16 input ---
    k_f32_to_f16<<<nb((long long)N_NODES * F_IN, 256), 256, 0, stream>>>(x, xb, N_NODES * F_IN);

    for (int L = 0; L < 3; ++L) {
        int K = (L == 0) ? F_IN : H_DIM;
        k_build_wT<<<nb((long long)COLS * K, 256), 256, 0, stream>>>(conv_root[L], conv_w[L], wT, K);
        dim3 grid(N_NODES / 16, 36 / 4);
        k_gemm_wmma<<<grid, 128, 0, stream>>>(xb, wT, h, K);
        k_acc_init<<<nb((long long)N_NODES * H_DIM, 256), 256, 0, stream>>>(h, conv_b[L], acc);
        k_scatter<<<nb((long long)N_EDGES * 32, 256), 256, 0, stream>>>(h, src, dst, et, ew, acc);
        k_zero_f<<<1, 128, 0, stream>>>(stats, 2 * H_DIM);
        k_colstats<<<256, 256, 0, stream>>>(acc, stats, stats + H_DIM);
        k_bn_coef<<<1, 64, 0, stream>>>(stats, stats + H_DIM, bn_g[L], bn_b[L],
                                        stats + 2 * H_DIM, stats + 3 * H_DIM);
        k_bn_relu<<<nb((long long)N_NODES * H_DIM, 256), 256, 0, stream>>>(
            acc, stats + 2 * H_DIM, stats + 3 * H_DIM, xl[L], xb);
    }

    // --- pooling ---
    k_zero_f<<<nb((long long)(G_GR * 192 * 2 + G_GR), 256), 256, 0, stream>>>(
        poolb, G_GR * 192 * 2 + G_GR);
    k_pool<<<nb((long long)N_NODES * 192, 256), 256, 0, stream>>>(
        xl[0], xl[1], xl[2], batch, psum, pmax, pcnt);
    k_xgraph<<<nb((long long)G_GR * 384, 256), 256, 0, stream>>>(psum, pmax, pcnt, xg);

    // --- head ---
    k_fc1<<<nb((long long)G_GR * 128, 256), 256, 0, stream>>>(xg, fc1_w, fc1_b, fcpre);
    k_fcbn_leaky<<<128, 256, 0, stream>>>(fcpre, fcbn_g, fcbn_b, fcb);
    k_fc2<<<nb((long long)G_GR * EMB_DIM, 256), 256, 0, stream>>>(fcb, fc2_w, fc2_b, emb_out);
    k_dec<<<nb(G_GR, 256), 256, 0, stream>>>(emb_out, dec1_w, dec1_b, dec2_w, dec2_b, pred_out);

    (void)in_sizes; (void)n_in; (void)out_size; (void)ws_size;
}
